// LocalPoolPointnet_24739011625527
// MI455X (gfx1250) — compile-verified, matrix operand order checked by reading, not run on hardware
//
#include <hip/hip_runtime.h>

typedef __attribute__((ext_vector_type(16))) __bf16 v16bf;
typedef __attribute__((ext_vector_type(16))) short  v16ss;
typedef __attribute__((ext_vector_type(8)))  float  v8f;
typedef __attribute__((ext_vector_type(4)))  unsigned uv4;

#define BATCH 4
#define NPTS  16384
#define MTOT  (BATCH * NPTS)
#define RESO  128
#define R2    (RESO * RESO)

// ---------- helpers ----------
// Involutive K-permutation (swap octets 1 and 2 in each 32-block). Activations
// and weights stored in this order make the ISA 16-bit A-fragment interleave a
// contiguous 32-byte load.
__device__ __forceinline__ int permc(int c) {
  int oct = (c >> 3) & 3;
  oct = (oct == 1) ? 2 : (oct == 2) ? 1 : oct;
  return (c & ~31) | (oct << 3) | (c & 7);
}

__device__ __forceinline__ unsigned short bf16_bits(float f) {
  union { __bf16 b; unsigned short u; } c;
  c.b = (__bf16)f;
  return c.u;
}

union Frag {
  uv4 u[2];
  v16bf v;
  v16ss s;
};

__device__ __forceinline__ v16bf frag_load(const unsigned short* p) {
  Frag f;
  f.u[0] = *(const uv4*)p;
  f.u[1] = *(const uv4*)(p + 8);
  return f.v;
}

// packed relu: zero every bf16 element whose sign bit is set
__device__ __forceinline__ v16bf relu16(v16bf a) {
  Frag f;
  f.v = a;
  v16ss m = f.s >> 15;      // 0xFFFF where negative
  f.s = f.s & ~m;
  return f.v;
}

__device__ __forceinline__ v8f wmma_bf16(v16bf a, v16bf b, v8f c) {
  return __builtin_amdgcn_wmma_f32_16x16x32_bf16(false, a, false, b, (short)0, c, false, false);
}

// ---------- utility kernels ----------
__global__ __launch_bounds__(256) void fill_u32_kernel(unsigned* __restrict__ p, unsigned v, long n) {
  long t = (long)blockIdx.x * 256 + threadIdx.x;
  if (t < n) p[t] = v;
}

// f32 weights -> bf16, permuting the K (input-feature) dimension
__global__ __launch_bounds__(256) void cvt_permK_kernel(const float* __restrict__ src,
                                                        unsigned short* __restrict__ dst,
                                                        int K, int n) {
  int t = blockIdx.x * 256 + threadIdx.x;
  if (t < n) {
    int col = t / K;
    int p = t - col * K;
    dst[t] = bf16_bits(src[col * K + permc(p)]);
  }
}

// ---------- positional encoding: idx, counts, net0 (bf16, K-permuted) ----------
__global__ __launch_bounds__(256) void pos_kernel(const float* __restrict__ pts,
                                                  const float* __restrict__ w,      // (256,3)
                                                  const float* __restrict__ bias,   // (256)
                                                  int* __restrict__ idx,
                                                  float* __restrict__ cnt,
                                                  unsigned short* __restrict__ net0) {
  int t = blockIdx.x * 256 + threadIdx.x;   // t = m*64 + q
  int m = t >> 6;
  int q = t & 63;
  float p0 = pts[3 * m + 0];
  float p1 = pts[3 * m + 1];
  float p2 = pts[3 * m + 2];
  if (q == 0) {
    int gx = (int)(p0 * (float)RESO); gx = gx < 0 ? 0 : (gx > RESO - 1 ? RESO - 1 : gx);
    int gy = (int)(p1 * (float)RESO); gy = gy < 0 ? 0 : (gy > RESO - 1 ? RESO - 1 : gy);
    int id = gx + RESO * gy;
    idx[m] = id;
    int b = m / NPTS;
    atomicAdd(&cnt[b * R2 + id], 1.0f);
  }
#pragma unroll
  for (int j = 0; j < 4; ++j) {
    int c = q * 4 + j;
    float v = p0 * w[c * 3 + 0] + p1 * w[c * 3 + 1] + p2 * w[c * 3 + 2] + bias[c];
    net0[(long)m * 256 + permc(c)] = bf16_bits(v);
  }
}

// ---------- segment-max scatter (integer-only keys) ----------
__global__ __launch_bounds__(256) void scatter_max_kernel(const unsigned short* __restrict__ net,
                                                          const int* __restrict__ idx,
                                                          unsigned* __restrict__ bins) {
  long t = (long)blockIdx.x * 256 + threadIdx.x;   // t < MTOT*128
  int c = (int)(t & 127);
  long m = t >> 7;
  unsigned bits = ((unsigned)net[m * 128 + c]) << 16;   // exact f32 pattern of the bf16
  unsigned key = (bits & 0x80000000u) ? ~bits : (bits | 0x80000000u);
  int b = (int)(m / NPTS);
  long row = (long)b * R2 + idx[m];
  atomicMax(&bins[row * 128 + c], key);
}

// keys -> bf16 pooled values (empty bins decode to NaN; never gathered)
__global__ __launch_bounds__(256) void decode_bins_kernel(const unsigned* __restrict__ bins,
                                                          unsigned short* __restrict__ outbf,
                                                          long n) {
  long t = (long)blockIdx.x * 256 + threadIdx.x;
  if (t < n) {
    unsigned k = bins[t];
    unsigned bits = (k & 0x80000000u) ? (k ^ 0x80000000u) : ~k;
    outbf[t] = (unsigned short)(bits >> 16);
  }
}

// ---------- fused ResnetBlockFC (bf16 WMMA) ----------
// out = x @ wsc^T + relu( relu(x) @ w0^T + b0 ) @ w1^T + b1
// x = [x1 (128) | pooled/x2 (128)] in K-permuted bf16 layout.
// Each wave owns 16 rows x 128 cols; A fragments load straight from global;
// x is read exactly once (relu'd copy feeds w0, raw copy feeds wsc).
// Only LDS use: per-wave transpose of h (C-layout -> A-layout).
__global__ __launch_bounds__(256) void resblock_kernel(
    const unsigned short* __restrict__ x1, int x1stride,
    const unsigned short* __restrict__ x2, int x2stride,       // k==0 second half
    const unsigned short* __restrict__ poolbf,                 // k>0 decoded bins
    const int*            __restrict__ idx,
    const unsigned short* __restrict__ w0, const float* __restrict__ b0,   // (128,256)
    const unsigned short* __restrict__ w1, const float* __restrict__ b1,   // (128,128)
    const unsigned short* __restrict__ wsc,                                 // (128,256)
    unsigned short* __restrict__ outp)
{
  __shared__ __align__(16) unsigned short hs[8][16][136];
  const int tid  = threadIdx.x;
  const int lane = tid & 31;
  const int wv   = tid >> 5;                 // 0..7
  const long m0  = (long)blockIdx.x * 128;
  const long mw0 = m0 + wv * 16;             // this wave's row base
  const int lrow  = lane & 15;
  const int hsel  = (lane >> 4) * 16;        // K sub-window offset
  const int rbase = hsel >> 1;               // 0 or 8 (C/D row offset)

  const long m = mw0 + lrow;                 // A row owned by this lane
  const unsigned short* arow1 = x1 + m * x1stride;
  const unsigned short* arow2;
  if (idx) {
    const long bbase = (m0 / NPTS) * (long)R2;
    arow2 = poolbf + (bbase + idx[m]) * 128;
  } else {
    arow2 = x2 + m * x2stride;
  }

  // ---- merged: acc = relu(x)@w0^T, acc2 = x@wsc^T ----
  v8f acc[8] = {};
  v8f acc2[8] = {};
#pragma unroll
  for (int kt = 0; kt < 8; ++kt) {
    const unsigned short* ap = (kt < 4) ? (arow1 + kt * 32 + hsel)
                                        : (arow2 + (kt - 4) * 32 + hsel);
    const v16bf araw = frag_load(ap);
    const v16bf arel = relu16(araw);
    const int kb = kt * 32 + hsel;
#pragma unroll
    for (int ns = 0; ns < 8; ++ns) {
      const int col = ns * 16 + lrow;
      acc[ns]  = wmma_bf16(arel, frag_load(w0  + (long)col * 256 + kb), acc[ns]);
      acc2[ns] = wmma_bf16(araw, frag_load(wsc + (long)col * 256 + kb), acc2[ns]);
    }
  }

  // ---- h = relu(acc + b0) -> LDS (per-wave slab, permuted col) ----
#pragma unroll
  for (int ns = 0; ns < 8; ++ns) {
    const int col = ns * 16 + lrow;
    const int pc = permc(col);
    const float bias = b0[col];
#pragma unroll
    for (int r = 0; r < 8; ++r) {
      const float v = acc[ns][r] + bias;
      hs[wv][rbase + r][pc] = bf16_bits(v > 0.f ? v : 0.f);
    }
  }

  // ---- acc2 += relu(h) @ w1^T (same-wave LDS, in-order DS) ----
#pragma unroll
  for (int kt = 0; kt < 4; ++kt) {
    const int kb = kt * 32 + hsel;
    const v16bf a = frag_load(&hs[wv][lrow][kb]);
#pragma unroll
    for (int ns = 0; ns < 8; ++ns) {
      const int col = ns * 16 + lrow;
      acc2[ns] = wmma_bf16(a, frag_load(w1 + (long)col * 128 + kb), acc2[ns]);
    }
  }

  // ---- store net_out (bf16, K-permuted feature order) ----
#pragma unroll
  for (int ns = 0; ns < 8; ++ns) {
    const int col = ns * 16 + lrow;
    const int pc = permc(col);
    const float bias = b1[col];
#pragma unroll
    for (int r = 0; r < 8; ++r) {
      outp[(mw0 + rbase + r) * 128 + pc] = bf16_bits(acc2[ns][r] + bias);
    }
  }
}

// ---------- head: relu(net) @ fc_c^T + b, scatter-add into plane sums ----------
__global__ __launch_bounds__(256) void head_kernel(const unsigned short* __restrict__ net,
                                                   const unsigned short* __restrict__ wc,  // (128,128) K-perm
                                                   const float* __restrict__ bc,
                                                   const int* __restrict__ idx,
                                                   float* __restrict__ sums) {
  const int tid  = threadIdx.x;
  const int lane = tid & 31;
  const int wv   = tid >> 5;
  const long m0  = (long)blockIdx.x * 128;
  const long mw0 = m0 + wv * 16;
  const int lrow  = lane & 15;
  const int hsel  = (lane >> 4) * 16;
  const int rbase = hsel >> 1;

  const unsigned short* arow = net + (mw0 + lrow) * 128;

  v8f acc[8] = {};
#pragma unroll
  for (int kt = 0; kt < 4; ++kt) {
    const int kb = kt * 32 + hsel;
    const v16bf a = relu16(frag_load(arow + kb));
#pragma unroll
    for (int ns = 0; ns < 8; ++ns) {
      const int col = ns * 16 + lrow;
      acc[ns] = wmma_bf16(a, frag_load(wc + (long)col * 128 + kb), acc[ns]);
    }
  }

  const long bbase = (m0 / NPTS) * (long)R2;
  long binrow[8];
#pragma unroll
  for (int r = 0; r < 8; ++r) binrow[r] = bbase + idx[mw0 + rbase + r];

#pragma unroll
  for (int ns = 0; ns < 8; ++ns) {
    const int col = ns * 16 + lrow;
    const float bias = bc[col];
#pragma unroll
    for (int r = 0; r < 8; ++r) {
      atomicAdd(&sums[binrow[r] * 128 + col], acc[ns][r] + bias);
    }
  }
}

// ---------- finalize: mean + transpose to (B, CDIM, RESO, RESO) ----------
__global__ __launch_bounds__(256) void finalize_kernel(const float* __restrict__ sums,
                                                       const float* __restrict__ cnt,
                                                       float* __restrict__ out) {
  long t = (long)blockIdx.x * 256 + threadIdx.x;  // t < BATCH*R2*128
  int c = (int)(t & 127);
  long br = t >> 7;                 // b*R2 + r
  long b = br >> 14;
  long r = br & (R2 - 1);
  float cv = cnt[br];
  float v = sums[br * 128 + c] / (cv > 1.f ? cv : 1.f);
  out[b * (128L * R2) + (long)c * R2 + r] = v;
}

// ---------- host ----------
extern "C" void kernel_launch(void* const* d_in, const int* in_sizes, int n_in,
                              void* d_out, int out_size, void* d_ws, size_t ws_size,
                              hipStream_t stream) {
  (void)in_sizes; (void)n_in; (void)out_size; (void)ws_size;
  const float* points    = (const float*)d_in[0];
  const float* fc_pos_w  = (const float*)d_in[1];
  const float* fc_pos_b  = (const float*)d_in[2];
  const float* blk_fc0_w = (const float*)d_in[3];
  const float* blk_fc0_b = (const float*)d_in[4];
  const float* blk_fc1_w = (const float*)d_in[5];
  const float* blk_fc1_b = (const float*)d_in[6];
  const float* blk_sc_w  = (const float*)d_in[7];
  const float* fc_c_w    = (const float*)d_in[8];
  const float* fc_c_b    = (const float*)d_in[9];

  char* wsb = (char*)d_ws;
  size_t off = 0;
  auto alloc = [&](size_t bytes) -> void* {
    void* p = wsb + off;
    off += (bytes + 255) & ~(size_t)255;
    return p;
  };

  float*          sums   = (float*)alloc((size_t)BATCH * R2 * 128 * 4);  // cnt follows contiguously
  float*          cnt    = (float*)alloc((size_t)BATCH * R2 * 4);
  int*            idxb   = (int*)alloc((size_t)MTOT * 4);
  unsigned short* w0a    = (unsigned short*)alloc((size_t)5 * 128 * 256 * 2);
  unsigned short* w1a    = (unsigned short*)alloc((size_t)5 * 128 * 128 * 2);
  unsigned short* wsca   = (unsigned short*)alloc((size_t)5 * 128 * 256 * 2);
  unsigned short* wcc    = (unsigned short*)alloc((size_t)128 * 128 * 2);
  unsigned short* net0   = (unsigned short*)alloc((size_t)MTOT * 256 * 2);
  unsigned short* netA   = (unsigned short*)alloc((size_t)MTOT * 128 * 2);
  unsigned short* netB   = (unsigned short*)alloc((size_t)MTOT * 128 * 2);
  unsigned*       bins   = (unsigned*)alloc((size_t)BATCH * R2 * 128 * 4);
  unsigned short* binsbf = (unsigned short*)alloc((size_t)BATCH * R2 * 128 * 2);

  // zero sums + cnt (contiguous)
  {
    long n = (long)BATCH * R2 * 128 + (long)BATCH * R2;
    fill_u32_kernel<<<(int)((n + 255) / 256), 256, 0, stream>>>((unsigned*)sums, 0u, n);
  }
  // weights -> bf16, K-permuted
  cvt_permK_kernel<<<(5 * 128 * 256 + 255) / 256, 256, 0, stream>>>(blk_fc0_w, w0a, 256, 5 * 128 * 256);
  cvt_permK_kernel<<<(5 * 128 * 128 + 255) / 256, 256, 0, stream>>>(blk_fc1_w, w1a, 128, 5 * 128 * 128);
  cvt_permK_kernel<<<(5 * 128 * 256 + 255) / 256, 256, 0, stream>>>(blk_sc_w, wsca, 256, 5 * 128 * 256);
  cvt_permK_kernel<<<(128 * 128 + 255) / 256, 256, 0, stream>>>(fc_c_w, wcc, 128, 128 * 128);

  // idx, counts, net0
  pos_kernel<<<MTOT * 64 / 256, 256, 0, stream>>>(points, fc_pos_w, fc_pos_b, idxb, cnt, net0);

  // block 0: dense 256-wide input
  resblock_kernel<<<MTOT / 128, 256, 0, stream>>>(net0, 256, net0 + 128, 256, nullptr, nullptr,
                                                  w0a, blk_fc0_b, w1a, blk_fc1_b, wsca, netA);

  unsigned short* cur = netA;
  unsigned short* nxt = netB;
  const long binsN = (long)BATCH * R2 * 128;
  for (int k = 1; k < 5; ++k) {
    fill_u32_kernel<<<(int)((binsN + 255) / 256), 256, 0, stream>>>(bins, 0u, binsN);
    scatter_max_kernel<<<(int)(((long)MTOT * 128) / 256), 256, 0, stream>>>(cur, idxb, bins);
    decode_bins_kernel<<<(int)((binsN + 255) / 256), 256, 0, stream>>>(bins, binsbf, binsN);
    resblock_kernel<<<MTOT / 128, 256, 0, stream>>>(cur, 128, nullptr, 0, binsbf, idxb,
                                                    w0a + (long)k * 128 * 256, blk_fc0_b + k * 128,
                                                    w1a + (long)k * 128 * 128, blk_fc1_b + k * 128,
                                                    wsca + (long)k * 128 * 256, nxt);
    unsigned short* t = cur; cur = nxt; nxt = t;
  }

  head_kernel<<<MTOT / 128, 256, 0, stream>>>(cur, wcc, fc_c_b, idxb, sums);
  finalize_kernel<<<(int)(((long)BATCH * R2 * 128) / 256), 256, 0, stream>>>(sums, cnt, (float*)d_out);
}
